// Model_71846212928220
// MI455X (gfx1250) — matrix operand order repaired and checked by reading, not compiled, not run on hardware
//
#include <hip/hip_runtime.h>
#include <hip/hip_bf16.h>

// ---------------------------------------------------------------------------
// GALA autoencoder: 6 x (dense + masked-adjacency aggregate + relu)
// B=1024, N=255 (padded to 256), channel dims padded to multiples of 32.
// All GEMMs via v_wmma_f32_16x16x32_bf16 (bf16 inputs, f32 accumulate).
// ---------------------------------------------------------------------------

typedef __attribute__((ext_vector_type(16))) __bf16 v16bf;
typedef __attribute__((ext_vector_type(8)))  __bf16 v8bf;
typedef __attribute__((ext_vector_type(8)))  float  v8f;

#define GALA_B   1024
#define GALA_NN  255
#define GALA_NP  256

// --- small prep kernels ----------------------------------------------------

// W [din x dout] f32 row-major  ->  Wt [doutp x dinp] bf16 (transposed, padded)
__global__ void wt_prep_kernel(const float* __restrict__ W, __bf16* __restrict__ Wt,
                               int din, int dout, int dinp, int doutp) {
    int idx = blockIdx.x * 256 + threadIdx.x;
    if (idx >= dinp * doutp) return;
    int c = idx / dinp;
    int k = idx - c * dinp;
    float v = (c < dout && k < din) ? W[k * dout + c] : 0.f;
    Wt[(size_t)c * dinp + k] = (__bf16)v;
}

// H [B x 255 x 2] f32 -> Xb [B x 256 x 32] bf16 (zero padded)
__global__ void x0_prep_kernel(const float* __restrict__ H, __bf16* __restrict__ X) {
    int idx = blockIdx.x * 256 + threadIdx.x;   // B*256*32 total
    int k = idx & 31;
    int n = (idx >> 5) & 255;
    int b = idx >> 13;
    float v = 0.f;
    if (n < GALA_NN && k < 2) v = H[((size_t)b * GALA_NN + n) * 2 + k];
    X[idx] = (__bf16)v;
}

// --- dense: Yt[b][col][k] = bf16( X[b]@W + bias ) (transposed output) ------

__global__ void __launch_bounds__(256)
dense_kernel(const __bf16* __restrict__ X, const __bf16* __restrict__ Wt,
             const float* __restrict__ bias, __bf16* __restrict__ Yt,
             int dinp, int doutp, int dout) {
    const int rowTile = blockIdx.x;           // 0 .. B*16-1
    const int b  = rowTile >> 4;
    const int n0 = (rowTile & 15) * 16;
    const int lane = threadIdx.x & 31;
    const int wave = threadIdx.x >> 5;
    const int half = lane >> 4;               // which 16-lane half
    const int l16  = lane & 15;

    __shared__ __align__(32) __bf16 Alds[16 * 416];   // 16 rows x dinp (<=416)

    // stage A panel (rows n0..n0+15 of X[b], contiguous in global) into LDS
    {
        const v8bf* src = (const v8bf*)(X + ((size_t)(b * GALA_NP + n0)) * dinp);
        v8bf* dst = (v8bf*)Alds;
        const int nchunks = (16 * dinp) / 8;
        for (int e = threadIdx.x; e < nchunks; e += 256) dst[e] = src[e];
    }
    __syncthreads();

    const int ntiles = doutp >> 4;
    for (int t = wave; t < ntiles; t += 8) {
        const int col = t * 16 + l16;
        const __bf16* Bp = Wt + (size_t)col * dinp;
        v8f acc = {};
        for (int k0 = 0; k0 < dinp; k0 += 32) {
            v8bf alo = *(const v8bf*)&Alds[l16 * dinp + k0 + 16 * half];
            v8bf ahi = *(const v8bf*)&Alds[l16 * dinp + k0 + 16 * half + 8];
            v16bf a = __builtin_shufflevector(alo, ahi,
                        0,1,2,3,4,5,6,7,8,9,10,11,12,13,14,15);
            v16bf bb = *(const v16bf*)&Bp[k0 + 16 * half];
            acc = __builtin_amdgcn_wmma_f32_16x16x32_bf16(
                      false, a, false, bb, (short)0, acc, false, false);
        }
        float bv = (col < dout) ? bias[col] : 0.f;
        v8bf ov;
#pragma unroll
        for (int v = 0; v < 8; ++v) ov[v] = (__bf16)(acc[v] + bv);
        // D tile: lane col fixed, rows n0 + v + 8*half -> contiguous in Yt
        __bf16* Op = Yt + ((size_t)(b * doutp + col)) * GALA_NP + n0 + 8 * half;
        *(v8bf*)Op = ov;
    }
}

// --- aggregate: Z[b] = relu( (dyna[b] .* dad) @ Y[b] ) ---------------------

__global__ void __launch_bounds__(256)
agg_kernel(const __bf16* __restrict__ Yt, const float* __restrict__ dyna,
           const float* __restrict__ dad, __bf16* __restrict__ Z,
           float* __restrict__ out, int doutp, int dout, int last) {
    const int rowTile = blockIdx.x;           // 0 .. B*16-1
    const int b  = rowTile >> 4;
    const int i0 = (rowTile & 15) * 16;
    const int lane = threadIdx.x & 31;
    const int wave = threadIdx.x >> 5;
    const int half = lane >> 4;
    const int l16  = lane & 15;

    __shared__ __align__(32) __bf16 Mlds[16 * 256];   // masked adjacency tile

    // build bf16 mask tile: M[r][j] = dyna[b, i0+r, j] * dad[i0+r, j]
    {
        int r  = threadIdx.x >> 4;
        int j0 = (threadIdx.x & 15) * 16;
        int grow = i0 + r;
#pragma unroll
        for (int j = 0; j < 16; ++j) {
            int jj = j0 + j;
            float v = 0.f;
            if (grow < GALA_NN && jj < GALA_NN)
                v = dyna[((size_t)b * GALA_NN + grow) * GALA_NN + jj] *
                    dad[grow * GALA_NN + jj];
            Mlds[r * 256 + jj] = (__bf16)v;
        }
    }
    __syncthreads();

    const int ntiles = doutp >> 4;
    for (int t = wave; t < ntiles; t += 8) {
        const int col = t * 16 + l16;
        const __bf16* Bp = Yt + ((size_t)(b * doutp + col)) * GALA_NP;
        v8f acc = {};
#pragma unroll
        for (int k0 = 0; k0 < 256; k0 += 32) {
            v8bf alo = *(const v8bf*)&Mlds[l16 * 256 + k0 + 16 * half];
            v8bf ahi = *(const v8bf*)&Mlds[l16 * 256 + k0 + 16 * half + 8];
            v16bf a = __builtin_shufflevector(alo, ahi,
                        0,1,2,3,4,5,6,7,8,9,10,11,12,13,14,15);
            v16bf bb = *(const v16bf*)&Bp[k0 + 16 * half];
            acc = __builtin_amdgcn_wmma_f32_16x16x32_bf16(
                      false, a, false, bb, (short)0, acc, false, false);
        }
#pragma unroll
        for (int v = 0; v < 8; ++v) {
            float r = acc[v] > 0.f ? acc[v] : 0.f;
            int n = i0 + v + 8 * half;
            if (!last) {
                Z[((size_t)(b * GALA_NP) + n) * doutp + col] = (__bf16)r;
            } else if (col < 2 && n < GALA_NN) {
                out[((size_t)b * GALA_NN + n) * 2 + col] = r;
            }
        }
    }
}

// ---------------------------------------------------------------------------

extern "C" void kernel_launch(void* const* d_in, const int* in_sizes, int n_in,
                              void* d_out, int out_size, void* d_ws, size_t ws_size,
                              hipStream_t stream) {
    static const int DIN[6]   = {2, 400, 300, 100, 300, 400};
    static const int DOUT[6]  = {400, 300, 100, 300, 400, 2};
    static const int DINP[6]  = {32, 416, 320, 128, 320, 416};
    static const int DOUTP[6] = {416, 320, 128, 320, 416, 32};

    const float* H    = (const float*)d_in[0];
    const float* dyna = (const float*)d_in[1];
    const float* dadp[6] = {
        (const float*)d_in[2],   // t_DADsm
        (const float*)d_in[2],   // t_DADsm
        (const float*)d_in[3],   // s_DADsm
        (const float*)d_in[4],   // s_DADsp
        (const float*)d_in[5],   // t_DADsp
        (const float*)d_in[5],   // t_DADsp
    };
    const float* W[6];
    const float* bias[6];
    for (int l = 0; l < 6; ++l) {
        W[l]    = (const float*)d_in[6 + 2 * l];
        bias[l] = (const float*)d_in[7 + 2 * l];
    }

    char* ws = (char*)d_ws;
    const size_t BX = (size_t)GALA_B * GALA_NP * 416 * sizeof(__bf16);  // ~218 MB
    __bf16* Xb = (__bf16*)ws;
    __bf16* Yt = (__bf16*)(ws + BX);
    __bf16* Wt[6];
    size_t off = 2 * BX;
    for (int l = 0; l < 6; ++l) {
        Wt[l] = (__bf16*)(ws + off);
        off += (size_t)DINP[l] * DOUTP[l] * sizeof(__bf16);
    }

    // weight transpose+pad to bf16
    for (int l = 0; l < 6; ++l) {
        int n = DINP[l] * DOUTP[l];
        wt_prep_kernel<<<(n + 255) / 256, 256, 0, stream>>>(
            W[l], Wt[l], DIN[l], DOUT[l], DINP[l], DOUTP[l]);
    }
    // initial input to bf16, padded
    x0_prep_kernel<<<(GALA_B * GALA_NP * 32) / 256, 256, 0, stream>>>(H, Xb);

    const int rowTiles = GALA_B * (GALA_NP / 16);   // 16384 workgroups
    for (int l = 0; l < 6; ++l) {
        dense_kernel<<<rowTiles, 256, 0, stream>>>(
            Xb, Wt[l], bias[l], Yt, DINP[l], DOUTP[l], DOUT[l]);
        agg_kernel<<<rowTiles, 256, 0, stream>>>(
            Yt, dyna, dadp[l], Xb, (float*)d_out, DOUTP[l], DOUT[l], l == 5 ? 1 : 0);
    }
}
